// ScaledDotProductAttention_3985729651642
// MI455X (gfx1250) — matrix operand order repaired and checked by reading, not compiled, not run on hardware
//
#include <hip/hip_runtime.h>
#include <hip/hip_bf16.h>

// MI455X / gfx1250, wave32. Fused attention fwd:
//   scores = QK^T*scale + prev, mask, softmax -> weights (written), out = weights @ V
// bf16 WMMA for both matmuls (f32 accumulate); scores tile resident in LDS.
// P is re-packed once into bf16 A-fragment order in LDS for the PV matmul.

typedef __attribute__((ext_vector_type(16))) __bf16 v16bf;
typedef __attribute__((ext_vector_type(8)))  float  v8f;
typedef __attribute__((ext_vector_type(4)))  unsigned v4u;

#define S_    2048
#define DH_   64
#define ROWS  16
#define CW    256           // column super-tile width
#define NST   (S_ / CW)     // 8 super-tiles
#define NT    256           // threads per block (8 wave32)

// LDS layout (bytes)
#define OFF_SC    0                        // f32 scores/exp [16][2048]      = 131072
#define OFF_STAGE 131072                   // bf16 B-fragment stage (K/V)    =  32768
#define OFF_PB    (131072 + 32768)         // bf16 P in A-fragment order     =  65536
#define OFF_QT    (OFF_PB + 65536)         // f32 Q tile [16][64]            =   4096
#define OFF_RS    (OFF_QT + 4096)          // f32 rowmax[16], rowinv[16]     =    128
#define SMEM_BYTES (OFF_RS + 128)

union AF { unsigned d[8]; v4u q[2]; __bf16 h[16]; v16bf v; };

__device__ __forceinline__ unsigned pk2(float a, float b) {
  union { unsigned u; __bf16 h[2]; } t;
  t.h[0] = (__bf16)a; t.h[1] = (__bf16)b;
  return t.u;
}

extern "C" __global__ __launch_bounds__(NT, 1)
void attn_fwd(const float* __restrict__ q, const float* __restrict__ k,
              const float* __restrict__ v, const float* __restrict__ prev,
              const unsigned char* __restrict__ mask, const float* __restrict__ scale_p,
              float* __restrict__ out, float* __restrict__ wts)
{
  extern __shared__ char smem[];
  float*    sc    = (float*)(smem + OFF_SC);
  unsigned* stage = (unsigned*)(smem + OFF_STAGE);
  unsigned* pbuf  = (unsigned*)(smem + OFF_PB);     // 64 fragments * 256 dwords
  float*    qt    = (float*)(smem + OFF_QT);
  float*    rmax  = (float*)(smem + OFF_RS);
  float*    rinv  = rmax + 16;

  const int tid  = threadIdx.x;
  const int lane = tid & 31;
  const int wave = tid >> 5;
  const int half = lane >> 4;   // which 16-lane half
  const int lm   = lane & 15;

  const int bh   = blockIdx.x >> 7;   // 128 row-tiles per (b,h)
  const int rt   = blockIdx.x & 127;
  const int b    = bh >> 4;           // H = 16
  const int row0 = rt * ROWS;

  const float scl = scale_p[0];
  const float* qb = q    + (size_t)bh * S_ * DH_;
  const float* kb = k    + (size_t)bh * DH_ * S_;   // k is [DH][S] per (b,h)
  const float* vb = v    + (size_t)bh * S_ * DH_;
  const float* pb = prev + (size_t)bh * S_ * S_;
  const unsigned char* mb = mask + (size_t)b * S_;
  float* wb = wts + (size_t)bh * S_ * S_;
  float* ob = out + (size_t)bh * S_ * DH_;

  // ---------- load Q tile (16x64 f32) ----------
  for (int i = tid; i < ROWS * DH_; i += NT) qt[i] = qb[(size_t)row0 * DH_ + i];
  __syncthreads();

  // ---------- per-wave Q A-fragments (ISA 16-bit A 16x32 layout) ----------
  // lane half h: e<8 -> K = h*8+e ; e>=8 -> K = 16 + h*8 + (e-8)
  AF aq0, aq1;
  #pragma unroll
  for (int e = 0; e < 16; ++e) {
    int dd = (e < 8) ? (half * 8 + e) : (16 + half * 8 + (e - 8));
    aq0.h[e] = (__bf16)qt[lm * DH_ + dd];
    aq1.h[e] = (__bf16)qt[lm * DH_ + 32 + dd];
  }

  // ================= QK^T * scale + prev, mask =================
  for (int st = 0; st < NST; ++st) {
    // stage K chunk (64 x 256) into B-fragment order, bf16.
    // B frag (32x16): lane l = hb*16+n holds e<8 -> K = hb*8+e ; e>=8 -> K = 16+hb*8+(e-8), col n.
    #pragma unroll
    for (int it = 0; it < 32; ++it) {
      int i   = it * NT + tid;
      int cl  = i & 255;           // col within super-tile (fastest -> coalesced)
      int dp  = i >> 8;            // d pair index 0..31
      int d   = dp * 2;
      int col = st * CW + cl;
      float f0 = kb[(size_t)d * S_ + col];
      float f1 = kb[(size_t)(d + 1) * S_ + col];
      int db = d >> 5, kk = d & 31, hb = (kk >> 3) & 1, e = ((kk >= 16) ? 8 : 0) + (kk & 7);
      int ct = cl >> 4, n = cl & 15, l = hb * 16 + n;
      stage[(ct * 2 + db) * 256 + l * 8 + (e >> 1)] = pk2(f0, f1);
    }
    __syncthreads();

    #pragma unroll
    for (int t = 0; t < 2; ++t) {
      int ct    = wave * 2 + t;
      int cbase = st * CW + ct * 16;
      AF b0, b1;
      {
        const v4u* bp0 = (const v4u*)(stage + (ct * 2 + 0) * 256 + lane * 8);
        const v4u* bp1 = (const v4u*)(stage + (ct * 2 + 1) * 256 + lane * 8);
        b0.q[0] = bp0[0]; b0.q[1] = bp0[1];   // ds_load_b128 x2
        b1.q[0] = bp1[0]; b1.q[1] = bp1[1];
      }
      v8f acc = {0.f,0.f,0.f,0.f,0.f,0.f,0.f,0.f};
      acc = __builtin_amdgcn_wmma_f32_16x16x32_bf16(false, aq0.v, false, b0.v,
                                                    (short)0, acc, false, false);
      acc = __builtin_amdgcn_wmma_f32_16x16x32_bf16(false, aq1.v, false, b1.v,
                                                    (short)0, acc, false, false);
      // D layout: VGPR r -> M = half*8 + r, N = lane&15
      #pragma unroll
      for (int r = 0; r < 8; ++r) {
        int m = half * 8 + r;
        int c = cbase + lm;
        float s = acc[r] * scl + pb[(size_t)(row0 + m) * S_ + c];
        s = mb[c] ? -__builtin_inff() : s;
        sc[m * S_ + c] = s;
      }
    }
    __syncthreads();
  }

  // ================= softmax stats (16 lanes per row) =================
  // Stores exp(s - rowmax) back into sc so exp is computed only once.
  {
    int r = tid >> 4, g = tid & 15;
    float mx = -__builtin_inff();
    for (int c = g; c < S_; c += 16) mx = fmaxf(mx, sc[r * S_ + c]);
    #pragma unroll
    for (int off = 8; off; off >>= 1) mx = fmaxf(mx, __shfl_xor(mx, off, 32));
    float sum = 0.f;
    for (int c = g; c < S_; c += 16) {
      float e = __expf(sc[r * S_ + c] - mx);
      sc[r * S_ + c] = e;
      sum += e;
    }
    #pragma unroll
    for (int off = 8; off; off >>= 1) sum += __shfl_xor(sum, off, 32);
    if (g == 0) { rmax[r] = mx; rinv[r] = 1.0f / sum; }
  }
  __syncthreads();

  // ======== normalize, write weights (float2), pack P bf16 A-fragments ========
  // Pairs of consecutive columns per thread. A-fragment addressing for pair at
  // even kk = c&31:  h = (kk>>3)&1, dword j = (kk>=16?4:0) + ((kk&7)>>1),
  // lane l = h*16 + row, fragment fb = c>>5.
  for (int i = tid; i < ROWS * (S_ / 2); i += NT) {
    int r  = i >> 10;          // 1024 pairs per row
    int cp = i & 1023;
    int c  = cp * 2;
    float iv = rinv[r];
    float p0 = sc[r * S_ + c]     * iv;
    float p1 = sc[r * S_ + c + 1] * iv;
    *(float2*)(wb + (size_t)(row0 + r) * S_ + c) = make_float2(p0, p1); // coalesced b64
    int kk = c & 31;
    int h  = (kk >> 3) & 1;
    int j  = ((kk >= 16) ? 4 : 0) + ((kk & 7) >> 1);
    int l  = h * 16 + r;
    pbuf[(c >> 5) * 256 + l * 8 + j] = pk2(p0, p1);
  }
  __syncthreads();

  // ================= out = P @ V =================
  const int tt = wave & 3;   // output 16-col tile of DH
  const int kh = wave >> 2;  // K-reduction half
  v8f oacc = {0.f,0.f,0.f,0.f,0.f,0.f,0.f,0.f};
  for (int stv = 0; stv < NST; ++stv) {
    // stage V chunk (256 x 64) into B-fragment order, bf16
    #pragma unroll
    for (int it = 0; it < 32; ++it) {
      int i  = it * NT + tid;
      int dh = i & 63;          // fastest -> coalesced over v's contiguous dim
      int cp = i >> 6;          // 0..127
      int cl = cp * 2;
      int c  = stv * CW + cl;
      float f0 = vb[(size_t)c * DH_ + dh];
      float f1 = vb[(size_t)(c + 1) * DH_ + dh];
      int cb = cl >> 5, kk = cl & 31, hb = (kk >> 3) & 1, e = ((kk >= 16) ? 8 : 0) + (kk & 7);
      int dt = dh >> 4, n = dh & 15, l = hb * 16 + n;
      stage[(cb * 4 + dt) * 256 + l * 8 + (e >> 1)] = pk2(f0, f1);
    }
    __syncthreads();

    #pragma unroll
    for (int j = 0; j < 4; ++j) {
      int cb = kh * 4 + j;
      int fb = stv * 8 + cb;       // P fragment index over full S
      AF ap, bv;
      {
        const v4u* app = (const v4u*)(pbuf + fb * 256 + lane * 8);
        const v4u* bpp = (const v4u*)(stage + (cb * 4 + tt) * 256 + lane * 8);
        ap.q[0] = app[0]; ap.q[1] = app[1];   // ds_load_b128 x2
        bv.q[0] = bpp[0]; bv.q[1] = bpp[1];
      }
      oacc = __builtin_amdgcn_wmma_f32_16x16x32_bf16(false, ap.v, false, bv.v,
                                                     (short)0, oacc, false, false);
    }
    __syncthreads();
  }

  // reduce the two K-half waves per output tile, write out
  float* red = (float*)stage;
  if (kh == 1) {
    #pragma unroll
    for (int r = 0; r < 8; ++r) red[tt * 256 + lane * 8 + r] = oacc[r];
  }
  __syncthreads();
  if (kh == 0) {
    #pragma unroll
    for (int r = 0; r < 8; ++r) {
      float o = oacc[r] + red[tt * 256 + lane * 8 + r];
      int m = half * 8 + r;
      ob[(size_t)(row0 + m) * DH_ + tt * 16 + lm] = o;
    }
  }
}

extern "C" void kernel_launch(void* const* d_in, const int* in_sizes, int n_in,
                              void* d_out, int out_size, void* d_ws, size_t ws_size,
                              hipStream_t stream) {
  (void)in_sizes; (void)n_in; (void)out_size; (void)d_ws; (void)ws_size;
  const float* q    = (const float*)d_in[0];
  const float* k    = (const float*)d_in[1];
  const float* v    = (const float*)d_in[2];
  const float* prev = (const float*)d_in[3];
  const unsigned char* mask = (const unsigned char*)d_in[4];  // jnp bool = 1 byte
  const float* scale = (const float*)d_in[5];

  float* out = (float*)d_out;                       // [B,H,S,DH] first
  float* wts = out + (size_t)2 * 16 * S_ * DH_;     // then [B,H,S,S]

  dim3 grid(2 * 16 * (S_ / ROWS));  // 4096 blocks
  attn_fwd<<<grid, NT, SMEM_BYTES, stream>>>(q, k, v, prev, mask, scale, out, wts);
}
